// CrossAttnV3_30193620091389
// MI455X (gfx1250) — matrix operand order repaired and checked by reading, not compiled, not run on hardware
//
#include <hip/hip_runtime.h>

// Problem constants (from reference)
#define B_    4
#define NQ_   1024
#define NKV_  2048
#define DQ_   1024
#define DC_   1024
#define DI_   1024
#define H_    16
#define DH_   64

// LDS tile leading dimension: 72 bf16 = 144B = 36 banks. For 16-lane fragment
// reads (stride ld*2B), bank start = 36n mod 64 -> {0,4,...,60}: conflict-free
// ds_load_b128, and rows stay 16B-aligned (144 = 9*16) for async b128 writes.
#define LDT   72

typedef __attribute__((ext_vector_type(16))) __bf16 v16bf;
typedef __attribute__((ext_vector_type(8)))  float  v8f;

union B16Frag { unsigned int u[8]; v16bf v; };

__device__ __forceinline__ unsigned short f2bf(float f) {
  unsigned int u = __float_as_uint(f);
  return (unsigned short)((u + 0x7FFFu + ((u >> 16) & 1u)) >> 16);  // RNE
}

__device__ __forceinline__ v8f wmma_bf16(v16bf a, v16bf b, v8f c) {
  return __builtin_amdgcn_wmma_f32_16x16x32_bf16(false, a, false, b, (short)0, c,
                                                 false, false);
}

// ---- async global -> LDS (CDNA5 ASYNCcnt path, inline asm) ------------------
// INST_OFFSET is added to BOTH the LDS and global addresses (ISA 10.x async),
// so one base pair + literal offsets covers a contiguous per-thread chunk.
#define ASYNC_B128(ldsoff, gptr, OFF)                                         \
  asm volatile("global_load_async_to_lds_b128 %0, %1, off offset:" OFF        \
               ::"v"(ldsoff), "v"(gptr)                                       \
               : "memory")

__device__ __forceinline__ unsigned lds_addr(const void* p) {
  return (unsigned)(unsigned long long)(uintptr_t)p;  // low 32 bits = LDS offset
}

__device__ __forceinline__ void async_copy_64B(unsigned lds, const void* g) {
  ASYNC_B128(lds, g, "0");
  ASYNC_B128(lds, g, "16");
  ASYNC_B128(lds, g, "32");
  ASYNC_B128(lds, g, "48");
}
__device__ __forceinline__ void async_copy_32B(unsigned lds, const void* g) {
  ASYNC_B128(lds, g, "0");
  ASYNC_B128(lds, g, "16");
}

__device__ __forceinline__ void wait_async_le8(int keep8) {
  if (keep8) asm volatile("s_wait_asynccnt 0x8" ::: "memory");
  else       asm volatile("s_wait_asynccnt 0x0" ::: "memory");
}
__device__ __forceinline__ void wait_async_le4(int keep4) {
  if (keep4) asm volatile("s_wait_asynccnt 0x4" ::: "memory");
  else       asm volatile("s_wait_asynccnt 0x0" ::: "memory");
}

// ---- fragment builders (CDNA5 ISA 7.12.2 layouts, wave32) -------------------
// A (16x32): lane row = l&15, h = l>>4; pair p: K = 2p + 8h (+8 for p>=4)
__device__ __forceinline__ v16bf fragA_lds(const unsigned short* tile, int rowBase,
                                           int ldk, int kBase) {
  const int l = threadIdx.x & 31, r = l & 15, h = l >> 4;
  const unsigned short* rp = tile + (rowBase + r) * ldk + kBase;
  B16Frag f;
#pragma unroll
  for (int p = 0; p < 8; ++p) {
    const int k0 = 2 * p + 8 * h + ((p >= 4) ? 8 : 0);
    f.u[p] = *(const unsigned int*)(rp + k0);
  }
  return f.v;
}

// B (32x16) from row-major [N][K] tile (C = A @ W^T): pair p: K = 2p + 8h (+8 for p>=4)
__device__ __forceinline__ v16bf fragB_nt_lds(const unsigned short* tile, int colBase,
                                              int ldk, int kBase) {
  const int l = threadIdx.x & 31, n = l & 15, h = l >> 4;
  const unsigned short* cp = tile + (colBase + n) * ldk + kBase;
  B16Frag f;
#pragma unroll
  for (int p = 0; p < 8; ++p) {
    const int k0 = 2 * p + 8 * h + ((p >= 4) ? 8 : 0);
    f.u[p] = *(const unsigned int*)(cp + k0);
  }
  return f.v;
}

// B (32x16) from row-major [K][N] tile (C = A @ Bmat)
__device__ __forceinline__ v16bf fragB_nn_lds(const unsigned short* tile, int kBase,
                                              int colBase, int ldn) {
  const int l = threadIdx.x & 31, n = l & 15, h = l >> 4;
  B16Frag f;
#pragma unroll
  for (int p = 0; p < 8; ++p) {
    const int k0 = kBase + 2 * p + 8 * h + ((p >= 4) ? 8 : 0);
    const unsigned int lo = tile[(k0)     * ldn + colBase + n];
    const unsigned int hi = tile[(k0 + 1) * ldn + colBase + n];
    f.u[p] = lo | (hi << 16);
  }
  return f.v;
}

// ---- LayerNorm -> bf16 ------------------------------------------------------
__global__ __launch_bounds__(256) void ln_bf16_kernel(
    const float* __restrict__ x, const float* __restrict__ gamma,
    const float* __restrict__ beta, unsigned short* __restrict__ out, int D) {
  __shared__ float sS[8], sQ[8];
  __shared__ float sMu, sRs;
  const int row = blockIdx.x;
  const float* xr = x + (size_t)row * D;
  float s = 0.f, q = 0.f;
  for (int i = threadIdx.x; i < D; i += 256) { float v = xr[i]; s += v; q += v * v; }
#pragma unroll
  for (int off = 16; off >= 1; off >>= 1) {
    s += __shfl_xor(s, off, 32);
    q += __shfl_xor(q, off, 32);
  }
  const int wave = threadIdx.x >> 5;
  if ((threadIdx.x & 31) == 0) { sS[wave] = s; sQ[wave] = q; }
  __syncthreads();
  if (threadIdx.x == 0) {
    float ts = 0.f, tq = 0.f;
#pragma unroll
    for (int i = 0; i < 8; ++i) { ts += sS[i]; tq += sQ[i]; }
    const float mu = ts / (float)D;
    const float var = tq / (float)D - mu * mu;
    sMu = mu;
    sRs = rsqrtf(var + 1e-5f);
  }
  __syncthreads();
  const float mu = sMu, rs = sRs;
  unsigned short* orow = out + (size_t)row * D;
  for (int i = threadIdx.x; i < D; i += 256)
    orow[i] = f2bf((xr[i] - mu) * rs * gamma[i] + beta[i]);
}

// ---- fp32 -> bf16 weight conversion ----------------------------------------
__global__ __launch_bounds__(256) void f32_to_bf16_kernel(
    const float* __restrict__ x, unsigned short* __restrict__ y, int n) {
  const int i = blockIdx.x * 256 + threadIdx.x;
  if (i < n) y[i] = f2bf(x[i]);
}

// ---- Tiled WMMA GEMM: C[b] = A[b] @ Bw^T (+bias) ----------------------------
// Block 128x128, 8 waves x (32x64), K-tile 64, 3-stage async->LDS pipeline.
// Per stage per thread: 8 async b128 ops (4 for A tile, 4 for B tile).
template <bool OUT_BF16, bool HAS_BIAS>
__global__ __launch_bounds__(256) void gemm_nt_kernel(
    const unsigned short* __restrict__ A, size_t aBatch, int lda,
    const unsigned short* __restrict__ Bw, int ldb,
    void* __restrict__ Cv, size_t cBatch, int ldc,
    const float* __restrict__ bias, int K) {
  __shared__ alignas(16) unsigned short lA[3][128 * LDT];  // 3 x 18KB
  __shared__ alignas(16) unsigned short lB[3][128 * LDT];  // 3 x 18KB
  const int t = threadIdx.x;
  const int wave = t >> 5, l = t & 31;
  const int waveM = wave & 3, waveN = wave >> 2;
  const int colBlock = blockIdx.x * 128;
  const int rowBlock = blockIdx.y * 128;
  const int b = blockIdx.z;
  const unsigned short* Ab = A + (size_t)b * aBatch;

  const v8f vzero = {};
  v8f acc[2][4];
#pragma unroll
  for (int mt = 0; mt < 2; ++mt)
#pragma unroll
    for (int nt = 0; nt < 4; ++nt) acc[mt][nt] = vzero;

  const int ldRow = t >> 1;        // 0..127
  const int ldCol = (t & 1) * 32;  // 0/32: 32 bf16 = 64B per thread per tile

  auto issue = [&](int stage, int k0) {
    const unsigned short* sa = Ab + (size_t)(rowBlock + ldRow) * lda + k0 + ldCol;
    async_copy_64B(lds_addr(&lA[stage][ldRow * LDT + ldCol]), sa);
    const unsigned short* sb = Bw + (size_t)(colBlock + ldRow) * ldb + k0 + ldCol;
    async_copy_64B(lds_addr(&lB[stage][ldRow * LDT + ldCol]), sb);
  };

  const int nIter = K >> 6;  // K / 64
  issue(0, 0);
  if (nIter > 1) issue(1, 64);

  for (int i = 0; i < nIter; ++i) {
    wait_async_le8(i + 1 < nIter);  // retire stage i; keep stage i+1 in flight
    __syncthreads();                // stage i visible to all waves
    const unsigned short* curA = lA[i % 3];
    const unsigned short* curB = lB[i % 3];
#pragma unroll
    for (int kk = 0; kk < 2; ++kk) {
      v16bf aF[2];
#pragma unroll
      for (int mt = 0; mt < 2; ++mt)
        aF[mt] = fragA_lds(curA, waveM * 32 + mt * 16, LDT, kk * 32);
#pragma unroll
      for (int nt = 0; nt < 4; ++nt) {
        const v16bf bF = fragB_nt_lds(curB, waveN * 64 + nt * 16, LDT, kk * 32);
#pragma unroll
        for (int mt = 0; mt < 2; ++mt) acc[mt][nt] = wmma_bf16(aF[mt], bF, acc[mt][nt]);
      }
    }
    if (i + 2 < nIter) issue((i + 2) % 3, (i + 2) * 64);
  }

  // D layout: VGPR r -> row (r + 8h), lanes 0..15 -> cols
  const int h = l >> 4, n16 = l & 15;
#pragma unroll
  for (int mt = 0; mt < 2; ++mt) {
#pragma unroll
    for (int nt = 0; nt < 4; ++nt) {
      const int row0 = rowBlock + waveM * 32 + mt * 16 + h * 8;
      const int col = colBlock + waveN * 64 + nt * 16 + n16;
#pragma unroll
      for (int r = 0; r < 8; ++r) {
        const float v = acc[mt][nt][r];
        if (OUT_BF16) {
          unsigned short* C = (unsigned short*)Cv + (size_t)b * cBatch;
          C[(size_t)(row0 + r) * ldc + col] = f2bf(v);
        } else {
          float* C = (float*)Cv + (size_t)b * cBatch;
          const float bv = HAS_BIAS ? bias[col] : 0.0f;
          C[(size_t)(row0 + r) * ldc + col] = v + bv;
        }
      }
    }
  }
}

// ---- Fused flash attention: softmax(Q K^T / sqrt(DH)) @ V -------------------
// grid = (NQ/128, H, B); wave owns a 16-row query strip; KV tiles of 64 via a
// 3-stage async->LDS pipeline (4 async b128 ops per thread per stage).
__global__ __launch_bounds__(256) void flash_attn_kernel(
    const unsigned short* __restrict__ qp,   // [B, NQ, DI] bf16
    const unsigned short* __restrict__ kv,   // [B, NKV, 2*DI] bf16 (K | V)
    unsigned short* __restrict__ aout) {     // [B, NQ, DI] bf16
  __shared__ alignas(16) unsigned short lK[3][64 * LDT];  // 3 x 9KB
  __shared__ alignas(16) unsigned short lV[3][64 * LDT];  // 3 x 9KB
  __shared__ alignas(16) unsigned short lP[8][16 * LDT];  // 18KB
  const int t = threadIdx.x;
  const int wave = t >> 5, l = t & 31;
  const int r16 = l & 15, h = l >> 4;
  const int head = blockIdx.y, b = blockIdx.z;
  const int rowBase = blockIdx.x * 128 + wave * 16;

  // Resident Q fragments (A layout), DH=64 -> 2 K-steps of 32
  const unsigned short* qrow =
      qp + ((size_t)(b * NQ_ + rowBase + r16)) * DI_ + head * DH_;
  v16bf aQ[2];
#pragma unroll
  for (int kk = 0; kk < 2; ++kk) {
    B16Frag f;
#pragma unroll
    for (int p = 0; p < 8; ++p) {
      const int k0 = 2 * p + 8 * h + ((p >= 4) ? 8 : 0);
      f.u[p] = *(const unsigned int*)(qrow + kk * 32 + k0);
    }
    aQ[kk] = f.v;
  }

  const v8f vzero = {};
  v8f O[4];
#pragma unroll
  for (int dt = 0; dt < 4; ++dt) O[dt] = vzero;
  float rmax[8], rsum[8];
#pragma unroll
  for (int r = 0; r < 8; ++r) { rmax[r] = -1e30f; rsum[r] = 0.f; }

  const int ldR = t >> 2, ldC = (t & 3) * 16;  // 64 rows x 4 threads x 32B

  auto issueKV = [&](int stage, int j) {
    const unsigned short* ks =
        kv + ((size_t)(b * NKV_ + j + ldR)) * (2 * DI_) + head * DH_ + ldC;
    async_copy_32B(lds_addr(&lK[stage][ldR * LDT + ldC]), ks);
    async_copy_32B(lds_addr(&lV[stage][ldR * LDT + ldC]), ks + DI_);
  };

  const int nIter = NKV_ / 64;  // 32
  issueKV(0, 0);
  issueKV(1, 64);

  for (int i = 0; i < nIter; ++i) {
    wait_async_le4(i + 1 < nIter);
    __syncthreads();
    const unsigned short* curK = lK[i % 3];
    const unsigned short* curV = lV[i % 3];

    // S = (Q K^T) strip: 4 N-tiles of 16
    v8f S[4];
#pragma unroll
    for (int nt = 0; nt < 4; ++nt) S[nt] = vzero;
#pragma unroll
    for (int nt = 0; nt < 4; ++nt)
#pragma unroll
      for (int kk = 0; kk < 2; ++kk) {
        const v16bf bK = fragB_nt_lds(curK, nt * 16, LDT, kk * 32);
        S[nt] = wmma_bf16(aQ[kk], bK, S[nt]);
      }

    // scale + per-row max (16-lane halves are rows r / r+8)
    float tmax[8];
#pragma unroll
    for (int r = 0; r < 8; ++r) {
      float m = -1e30f;
#pragma unroll
      for (int nt = 0; nt < 4; ++nt) {
        S[nt][r] *= 0.125f;  // DH^-0.5
        m = fmaxf(m, S[nt][r]);
      }
      tmax[r] = m;
    }
#pragma unroll
    for (int r = 0; r < 8; ++r)
#pragma unroll
      for (int off = 8; off >= 1; off >>= 1)
        tmax[r] = fmaxf(tmax[r], __shfl_xor(tmax[r], off, 16));

    float corr[8], tsum[8];
#pragma unroll
    for (int r = 0; r < 8; ++r) {
      const float nm = fmaxf(rmax[r], tmax[r]);
      corr[r] = __expf(rmax[r] - nm);
      rmax[r] = nm;
      float s = 0.f;
#pragma unroll
      for (int nt = 0; nt < 4; ++nt) {
        const float e = __expf(S[nt][r] - nm);
        S[nt][r] = e;
        s += e;
      }
      tsum[r] = s;
    }
#pragma unroll
    for (int r = 0; r < 8; ++r) {
#pragma unroll
      for (int off = 8; off >= 1; off >>= 1) tsum[r] += __shfl_xor(tsum[r], off, 16);
      rsum[r] = rsum[r] * corr[r] + tsum[r];
    }
#pragma unroll
    for (int dt = 0; dt < 4; ++dt)
#pragma unroll
      for (int r = 0; r < 8; ++r) O[dt][r] *= corr[r];

    // Stage P (D layout -> row major bf16) in per-wave LDS, re-read as A frags
    unsigned short* myP = lP[wave];
#pragma unroll
    for (int nt = 0; nt < 4; ++nt)
#pragma unroll
      for (int r = 0; r < 8; ++r)
        myP[(r + 8 * h) * LDT + nt * 16 + r16] = f2bf(S[nt][r]);
    asm volatile("s_wait_dscnt 0" ::: "memory");

    v16bf aP[2];
#pragma unroll
    for (int kk = 0; kk < 2; ++kk) aP[kk] = fragA_lds(myP, 0, LDT, kk * 32);

#pragma unroll
    for (int dt = 0; dt < 4; ++dt)
#pragma unroll
      for (int kk = 0; kk < 2; ++kk) {
        const v16bf bV = fragB_nn_lds(curV, kk * 32, dt * 16, LDT);
        O[dt] = wmma_bf16(aP[kk], bV, O[dt]);
      }

    if (i + 2 < nIter) issueKV((i + 2) % 3, (i + 2) * 64);
  }

  // O /= rowsum, store bf16
#pragma unroll
  for (int dt = 0; dt < 4; ++dt) {
    const int col = head * DH_ + dt * 16 + r16;
#pragma unroll
    for (int r = 0; r < 8; ++r) {
      const int row = rowBase + r + 8 * h;
      aout[((size_t)(b * NQ_ + row)) * DI_ + col] = f2bf(O[dt][r] / rsum[r]);
    }
  }
}

// ---- Launch -----------------------------------------------------------------
extern "C" void kernel_launch(void* const* d_in, const int* in_sizes, int n_in,
                              void* d_out, int out_size, void* d_ws, size_t ws_size,
                              hipStream_t stream) {
  (void)in_sizes; (void)n_in; (void)out_size; (void)ws_size;
  const float* q     = (const float*)d_in[0];
  const float* ctx   = (const float*)d_in[1];
  const float* Wq    = (const float*)d_in[2];
  const float* Wkv   = (const float*)d_in[3];
  const float* Wproj = (const float*)d_in[4];
  const float* bproj = (const float*)d_in[5];
  const float* qg    = (const float*)d_in[6];
  const float* qb    = (const float*)d_in[7];
  const float* cg    = (const float*)d_in[8];
  const float* cb    = (const float*)d_in[9];
  float* out = (float*)d_out;

  // Workspace layout (bf16), ~80 MB total
  unsigned short* qn   = (unsigned short*)d_ws;                 // B*NQ*DQ
  unsigned short* cn   = qn   + (size_t)B_ * NQ_ * DQ_;         // B*NKV*DC
  unsigned short* wqb  = cn   + (size_t)B_ * NKV_ * DC_;        // DI*DQ
  unsigned short* wkvb = wqb  + (size_t)DI_ * DQ_;              // 2*DI*DC
  unsigned short* wpb  = wkvb + (size_t)2 * DI_ * DC_;          // DQ*DI
  unsigned short* qpb  = wpb  + (size_t)DQ_ * DI_;              // B*NQ*DI
  unsigned short* kvb  = qpb  + (size_t)B_ * NQ_ * DI_;         // B*NKV*2*DI
  unsigned short* aob  = kvb  + (size_t)B_ * NKV_ * 2 * DI_;    // B*NQ*DI

  // 1) LayerNorms -> bf16
  ln_bf16_kernel<<<B_ * NQ_, 256, 0, stream>>>(q, qg, qb, qn, DQ_);
  ln_bf16_kernel<<<B_ * NKV_, 256, 0, stream>>>(ctx, cg, cb, cn, DC_);

  // 2) Weights -> bf16
  f32_to_bf16_kernel<<<(DI_ * DQ_) / 256, 256, 0, stream>>>(Wq, wqb, DI_ * DQ_);
  f32_to_bf16_kernel<<<(2 * DI_ * DC_) / 256, 256, 0, stream>>>(Wkv, wkvb, 2 * DI_ * DC_);
  f32_to_bf16_kernel<<<(DQ_ * DI_) / 256, 256, 0, stream>>>(Wproj, wpb, DQ_ * DI_);

  // 3) qp = qn @ Wq^T  (bf16 out)
  gemm_nt_kernel<true, false><<<dim3(DI_ / 128, NQ_ / 128, B_), 256, 0, stream>>>(
      qn, (size_t)NQ_ * DQ_, DQ_, wqb, DQ_, qpb, (size_t)NQ_ * DI_, DI_, nullptr, DQ_);

  // 4) kv = cn @ Wkv^T (bf16 out, K|V concatenated along columns)
  gemm_nt_kernel<true, false><<<dim3(2 * DI_ / 128, NKV_ / 128, B_), 256, 0, stream>>>(
      cn, (size_t)NKV_ * DC_, DC_, wkvb, DC_, kvb, (size_t)NKV_ * 2 * DI_, 2 * DI_,
      nullptr, DC_);

  // 5) fused attention
  flash_attn_kernel<<<dim3(NQ_ / 128, H_, B_), 256, 0, stream>>>(qpb, kvb, aob);

  // 6) out = attn @ Wproj^T + bproj (fp32 out)
  gemm_nt_kernel<false, true><<<dim3(DQ_ / 128, NQ_ / 128, B_), 256, 0, stream>>>(
      aob, (size_t)NQ_ * DI_, DI_, wpb, DI_, out, (size_t)NQ_ * DQ_, DQ_, bproj, DI_);
}
